// GNN_3504693313899
// MI455X (gfx1250) — matrix operand order repaired and checked by reading, not compiled, not run on hardware
//
#include <hip/hip_runtime.h>
#include <hip/hip_bf16.h>

typedef __attribute__((ext_vector_type(16))) __bf16 v16bf;
typedef __attribute__((ext_vector_type(8)))  __bf16 v8bf;
typedef __attribute__((ext_vector_type(8)))  float  v8f;

constexpr int kNodes = 50000;
constexpr int kEdges = 800000;
constexpr int kFin   = 128;
constexpr int kHid   = 128;
constexpr int kCls   = 4;
constexpr int kWPad  = 136;   // padded LDS row stride (bf16): 272 B, 16B-aligned, bank-spread

__device__ __forceinline__ float relu_f(float v) { return v > 0.0f ? v : 0.0f; }

// two runs of 8 bf16 from LDS (p[0..7], p[16..23]) as two ds_load_b128
__device__ __forceinline__ v16bf load_bfrag16(const __bf16* p) {
    v8bf lo = *(const v8bf*)(p);
    v8bf hi = *(const v8bf*)(p + 16);
    return __builtin_shufflevector(lo, hi, 0, 1, 2, 3, 4, 5, 6, 7,
                                           8, 9, 10, 11, 12, 13, 14, 15);
}

// ---------------------------------------------------------------- degree / norm
__global__ void k_deg_init(float* __restrict__ deg) {
    int n = blockIdx.x * 256 + threadIdx.x;
    if (n < kNodes) deg[n] = 1.0f;              // self-loop weight = 1
}

__global__ void k_deg_edges(const long long* __restrict__ dst,
                            const float* __restrict__ ew,
                            float* __restrict__ deg) {
    int e = blockIdx.x * 256 + threadIdx.x;
    if (e < kEdges) unsafeAtomicAdd(&deg[(int)dst[e]], ew[e]);
}

__global__ void k_dis(const float* __restrict__ deg, float* __restrict__ dis) {
    int n = blockIdx.x * 256 + threadIdx.x;
    if (n < kNodes) {
        float d = deg[n];
        dis[n] = (d > 0.0f) ? rsqrtf(d) : 0.0f;
    }
}

// ---------------------------------------------------------------- GEMM1
// xl = x @ W1 ; also emits h = xl * dis^2 (self-loop init of the aggregator).
// Block = 8 waves, 128 rows; wave = 16 rows; W1 in LDS transposed [n][k].
__global__ __launch_bounds__(256) void k_gemm1(const float* __restrict__ X,
                                               const float* __restrict__ W,
                                               const float* __restrict__ dis,
                                               float* __restrict__ XL,
                                               float* __restrict__ AGG) {
    __shared__ __bf16 sWt[kHid * kWPad];        // [n][k] transposed, ~34 KB
    for (int i = threadIdx.x; i < kFin * kHid; i += 256) {
        const int k = i >> 7, c = i & 127;
        sWt[c * kWPad + k] = (__bf16)W[i];
    }
    __syncthreads();

    const int wave = threadIdx.x >> 5;          // wave32
    const int lane = threadIdx.x & 31;
    const int hl   = lane >> 4;
    const int col  = lane & 15;
    const int rowBase = blockIdx.x * 128 + wave * 16;
    const bool full = (rowBase + 16 <= kNodes);

    int arow = rowBase + (lane & 15);
    if (arow >= kNodes) arow = kNodes - 1;      // clamp; masked on store
    const float* xb = X + (size_t)arow * kFin;

    // A fragments: per K-step, each lane needs two runs of 8 contiguous floats
    v16bf a[4];
#pragma unroll
    for (int ks = 0; ks < 4; ++ks) {
        const int k0 = ks * 32 + hl * 8;        // -> a[0..7]
        const int k1 = k0 + 16;                 // -> a[8..15]
        const float4 t0 = *(const float4*)(xb + k0);
        const float4 t1 = *(const float4*)(xb + k0 + 4);
        const float4 t2 = *(const float4*)(xb + k1);
        const float4 t3 = *(const float4*)(xb + k1 + 4);
        a[ks][0]  = (__bf16)t0.x; a[ks][1]  = (__bf16)t0.y;
        a[ks][2]  = (__bf16)t0.z; a[ks][3]  = (__bf16)t0.w;
        a[ks][4]  = (__bf16)t1.x; a[ks][5]  = (__bf16)t1.y;
        a[ks][6]  = (__bf16)t1.z; a[ks][7]  = (__bf16)t1.w;
        a[ks][8]  = (__bf16)t2.x; a[ks][9]  = (__bf16)t2.y;
        a[ks][10] = (__bf16)t2.z; a[ks][11] = (__bf16)t2.w;
        a[ks][12] = (__bf16)t3.x; a[ks][13] = (__bf16)t3.y;
        a[ks][14] = (__bf16)t3.z; a[ks][15] = (__bf16)t3.w;
    }

    // dis^2 per output row (rows rowBase + hl*8 + r)
    float d2[8];
#pragma unroll
    for (int r = 0; r < 8; ++r) {
        const int orow = rowBase + hl * 8 + r;
        const float dv = (orow < kNodes) ? dis[orow] : 0.0f;
        d2[r] = dv * dv;
    }

#pragma unroll
    for (int nt = 0; nt < 8; ++nt) {
        v8f acc = {};
#pragma unroll
        for (int ks = 0; ks < 4; ++ks) {
            // B fragment: two runs of 8 bf16 (K, 16 apart) at transposed row nt*16+col
            const v16bf b = load_bfrag16(&sWt[(nt * 16 + col) * kWPad + ks * 32 + hl * 8]);
            acc = __builtin_amdgcn_wmma_f32_16x16x32_bf16(
                false, a[ks], false, b, (short)0, acc, false, false);
        }
        if (full) {
#pragma unroll
            for (int r = 0; r < 8; ++r) {
                const size_t o = (size_t)(rowBase + hl * 8 + r) * kHid + nt * 16 + col;
                XL[o]  = acc[r];
                AGG[o] = acc[r] * d2[r];
            }
        } else {
#pragma unroll
            for (int r = 0; r < 8; ++r) {
                const int orow = rowBase + hl * 8 + r;
                if (orow < kNodes) {
                    const size_t o = (size_t)orow * kHid + nt * 16 + col;
                    XL[o]  = acc[r];
                    AGG[o] = acc[r] * d2[r];
                }
            }
        }
    }
}

// ---------------------------------------------------------------- edge scatter L1
__global__ __launch_bounds__(256) void k_edge1(const long long* __restrict__ src,
                                               const long long* __restrict__ dst,
                                               const float* __restrict__ ew,
                                               const float* __restrict__ dis,
                                               const float* __restrict__ xl,
                                               float* __restrict__ agg) {
    const int e = blockIdx.x * 2 + (threadIdx.x >> 7);   // 128 lanes per edge
    const int f = threadIdx.x & 127;
    if (e < kEdges) {
        const int s = (int)src[e];
        const int d = (int)dst[e];
        const float nrm = dis[s] * ew[e] * dis[d];
        unsafeAtomicAdd(&agg[(size_t)d * kHid + f], xl[(size_t)s * kHid + f] * nrm);
    }
}

// ---------------------------------------------------------------- GEMM2
// olin = relu(h + b1) @ W2 (N=4 padded to 16); also agg2 = olin * dis^2.
__global__ __launch_bounds__(256) void k_gemm2(const float* __restrict__ H,
                                               const float* __restrict__ W2,
                                               const float* __restrict__ b1,
                                               const float* __restrict__ dis,
                                               float* __restrict__ OL,
                                               float* __restrict__ AGG2) {
    __shared__ __bf16 sW2t[16 * kWPad];         // [n][k], cols 4..15 zero
    __shared__ float  sb1[kHid];
    for (int i = threadIdx.x; i < kHid * 16; i += 256) {
        const int k = i >> 4, c = i & 15;
        sW2t[c * kWPad + k] = (__bf16)((c < kCls) ? W2[k * kCls + c] : 0.0f);
    }
    if (threadIdx.x < kHid) sb1[threadIdx.x] = b1[threadIdx.x];
    __syncthreads();

    const int wave = threadIdx.x >> 5;
    const int lane = threadIdx.x & 31;
    const int hl   = lane >> 4;
    const int col  = lane & 15;
    const int rowBase = blockIdx.x * 128 + wave * 16;
    const bool full = (rowBase + 16 <= kNodes);

    int arow = rowBase + (lane & 15);
    if (arow >= kNodes) arow = kNodes - 1;
    const float* hb = H + (size_t)arow * kHid;

    v8f acc = {};
#pragma unroll
    for (int ks = 0; ks < 4; ++ks) {
        const int k0 = ks * 32 + hl * 8;
        const int k1 = k0 + 16;
        const float4 t0 = *(const float4*)(hb + k0);
        const float4 t1 = *(const float4*)(hb + k0 + 4);
        const float4 t2 = *(const float4*)(hb + k1);
        const float4 t3 = *(const float4*)(hb + k1 + 4);
        v16bf a;
        a[0]  = (__bf16)relu_f(t0.x + sb1[k0 + 0]);
        a[1]  = (__bf16)relu_f(t0.y + sb1[k0 + 1]);
        a[2]  = (__bf16)relu_f(t0.z + sb1[k0 + 2]);
        a[3]  = (__bf16)relu_f(t0.w + sb1[k0 + 3]);
        a[4]  = (__bf16)relu_f(t1.x + sb1[k0 + 4]);
        a[5]  = (__bf16)relu_f(t1.y + sb1[k0 + 5]);
        a[6]  = (__bf16)relu_f(t1.z + sb1[k0 + 6]);
        a[7]  = (__bf16)relu_f(t1.w + sb1[k0 + 7]);
        a[8]  = (__bf16)relu_f(t2.x + sb1[k1 + 0]);
        a[9]  = (__bf16)relu_f(t2.y + sb1[k1 + 1]);
        a[10] = (__bf16)relu_f(t2.z + sb1[k1 + 2]);
        a[11] = (__bf16)relu_f(t2.w + sb1[k1 + 3]);
        a[12] = (__bf16)relu_f(t3.x + sb1[k1 + 4]);
        a[13] = (__bf16)relu_f(t3.y + sb1[k1 + 5]);
        a[14] = (__bf16)relu_f(t3.z + sb1[k1 + 6]);
        a[15] = (__bf16)relu_f(t3.w + sb1[k1 + 7]);
        const v16bf b = load_bfrag16(&sW2t[col * kWPad + ks * 32 + hl * 8]);
        acc = __builtin_amdgcn_wmma_f32_16x16x32_bf16(
            false, a, false, b, (short)0, acc, false, false);
    }

    if (col < kCls) {
        if (full) {
#pragma unroll
            for (int r = 0; r < 8; ++r) {
                const int orow = rowBase + hl * 8 + r;
                const float dv = dis[orow];
                const size_t o = (size_t)orow * kCls + col;
                OL[o]   = acc[r];
                AGG2[o] = acc[r] * dv * dv;
            }
        } else {
#pragma unroll
            for (int r = 0; r < 8; ++r) {
                const int orow = rowBase + hl * 8 + r;
                if (orow < kNodes) {
                    const float dv = dis[orow];
                    const size_t o = (size_t)orow * kCls + col;
                    OL[o]   = acc[r];
                    AGG2[o] = acc[r] * dv * dv;
                }
            }
        }
    }
}

// ---------------------------------------------------------------- edge scatter L2 + softmax
__global__ void k_edge2(const long long* __restrict__ src, const long long* __restrict__ dst,
                        const float* __restrict__ ew, const float* __restrict__ dis,
                        const float* __restrict__ olin, float* __restrict__ agg2) {
    int e = blockIdx.x * 256 + threadIdx.x;
    if (e < kEdges) {
        const int s = (int)src[e];
        const int d = (int)dst[e];
        const float nrm = dis[s] * ew[e] * dis[d];
#pragma unroll
        for (int c = 0; c < kCls; ++c)
            unsafeAtomicAdd(&agg2[(size_t)d * kCls + c], olin[(size_t)s * kCls + c] * nrm);
    }
}

__global__ void k_softmax(const float* __restrict__ agg2, const float* __restrict__ b2,
                          float* __restrict__ out) {
    int n = blockIdx.x * 256 + threadIdx.x;
    if (n < kNodes) {
        float v[kCls];
        float m = -3.4e38f;
#pragma unroll
        for (int c = 0; c < kCls; ++c) {
            v[c] = agg2[n * kCls + c] + b2[c];
            m = fmaxf(m, v[c]);
        }
        float s = 0.0f;
#pragma unroll
        for (int c = 0; c < kCls; ++c) {
            v[c] = __expf(v[c] - m);
            s += v[c];
        }
        const float inv = 1.0f / s;
#pragma unroll
        for (int c = 0; c < kCls; ++c)
            out[n * kCls + c] = v[c] * inv;
    }
}

// ---------------------------------------------------------------- launcher
extern "C" void kernel_launch(void* const* d_in, const int* in_sizes, int n_in,
                              void* d_out, int out_size, void* d_ws, size_t ws_size,
                              hipStream_t stream) {
    (void)in_sizes; (void)n_in; (void)out_size; (void)ws_size;

    const float*     x   = (const float*)d_in[0];
    const long long* ei  = (const long long*)d_in[1];   // int64 per reference
    const float*     ew  = (const float*)d_in[2];
    const float*     W1  = (const float*)d_in[3];
    const float*     b1  = (const float*)d_in[4];
    const float*     W2  = (const float*)d_in[5];
    const float*     b2  = (const float*)d_in[6];
    const long long* src = ei;
    const long long* dst = ei + kEdges;
    float*           out = (float*)d_out;

    char* ws = (char*)d_ws;
    size_t off = 0;
    auto take = [&](size_t bytes) -> void* {
        void* p = ws + off;
        off += (bytes + 255) & ~(size_t)255;
        return p;
    };
    float* deg  = (float*)take((size_t)kNodes * sizeof(float));
    float* dis  = (float*)take((size_t)kNodes * sizeof(float));
    float* xl   = (float*)take((size_t)kNodes * kHid * sizeof(float));
    float* h    = (float*)take((size_t)kNodes * kHid * sizeof(float)); // aggregator L1
    float* olin = (float*)take((size_t)kNodes * kCls * sizeof(float));
    float* agg2 = (float*)take((size_t)kNodes * kCls * sizeof(float));

    const int nodeBlocks = (kNodes + 255) / 256;
    const int edgeBlocks = (kEdges + 255) / 256;
    const int gemmBlocks = (kNodes + 127) / 128;

    k_deg_init <<<nodeBlocks, 256, 0, stream>>>(deg);
    k_deg_edges<<<edgeBlocks, 256, 0, stream>>>(dst, ew, deg);
    k_dis      <<<nodeBlocks, 256, 0, stream>>>(deg, dis);

    k_gemm1    <<<gemmBlocks, 256, 0, stream>>>(x, W1, dis, xl, h);
    k_edge1    <<<(kEdges + 1) / 2, 256, 0, stream>>>(src, dst, ew, dis, xl, h);

    k_gemm2    <<<gemmBlocks, 256, 0, stream>>>(h, W2, b1, dis, olin, agg2);
    k_edge2    <<<edgeBlocks, 256, 0, stream>>>(src, dst, ew, dis, olin, agg2);
    k_softmax  <<<nodeBlocks, 256, 0, stream>>>(agg2, b2, out);
}
